// FAM_Deform_39230231281674
// MI455X (gfx1250) — matrix operand order repaired and checked by reading, not compiled, not run on hardware
//
#include <hip/hip_runtime.h>
#include <hip/hip_bf16.h>

// ---------------------------------------------------------------------------
// Deformable-conv fused pipeline for MI455X (gfx1250, wave32, WMMA).
//
// Shapes (from reference): B=4, C=64, H=W=160, K=3 -> K2=9, KC=C*K2=576.
//
// Design: keep the sampled [576]-vector per pixel in LDS only (saves ~236 MB
// of DRAM round-trip), run the 64x576 contraction on v_wmma_f32_16x16x32_bf16
// with f32 accumulation. All intermediates (x, offsets, bf16 weights) are
// L2-resident (192 MB L2 >> ~34 MB workspace).
// ---------------------------------------------------------------------------

#define BB 4
#define CC 64
#define HH 160
#define WW 160
#define K2 9
#define KC (CC * K2)      // 576
#define OFFC 18           // 2*K2 offset channels
#define TP 32             // pixels per block in the GEMM kernel
#define SROW (KC + 8)     // padded LDS row (584 elems) -> 16B aligned, spreads banks

typedef __attribute__((ext_vector_type(16))) __bf16 v16bf;
typedef __attribute__((ext_vector_type(8)))  __bf16 v8bf;
typedef __attribute__((ext_vector_type(8)))  float  v8f;

// ---------------------------------------------------------------- kernel 1 --
// x = x1 * x2, float4-vectorized (N = 6,553,600 elems = 1,638,400 float4)
__global__ __launch_bounds__(256) void mul_kernel(
    const float* __restrict__ x1, const float* __restrict__ x2,
    float* __restrict__ x, int n4) {
  int i = blockIdx.x * 256 + threadIdx.x;
  if (i < n4) {
    float4 a = ((const float4*)x1)[i];
    float4 b = ((const float4*)x2)[i];
    float4 r; r.x = a.x*b.x; r.y = a.y*b.y; r.z = a.z*b.z; r.w = a.w*b.w;
    ((float4*)x)[i] = r;
  }
}

// ---------------------------------------------------------------- kernel 2 --
// W_dc fp32 -> bf16, layout [O=64][KC=576] (same linear order as [O][C][3][3])
__global__ __launch_bounds__(256) void wconv_kernel(
    const float* __restrict__ wdc, __bf16* __restrict__ wbf, int n) {
  int i = blockIdx.x * 256 + threadIdx.x;
  if (i < n) wbf[i] = (__bf16)wdc[i];
}

// ---------------------------------------------------------------- kernel 3 --
// Offset conv: off[b][18][h][w] = b_off + sum_{c,ky,kx} W_off * x (SAME pad).
// Weights (18*64*9 fp32 = 41.5 KB) staged in LDS once per block.
__global__ __launch_bounds__(256) void offset_conv(
    const float* __restrict__ x, const float* __restrict__ woff,
    const float* __restrict__ boff, float* __restrict__ off) {
  __shared__ float sw[OFFC * CC * K2];
  for (int i = threadIdx.x; i < OFFC * CC * K2; i += 256) sw[i] = woff[i];
  __syncthreads();

  int gid = blockIdx.x * 256 + threadIdx.x;
  if (gid >= BB * HH * WW) return;
  int b  = gid / (HH * WW);
  int pr = gid % (HH * WW);
  int h  = pr / WW, w = pr % WW;

  float acc[OFFC];
#pragma unroll
  for (int o = 0; o < OFFC; ++o) acc[o] = boff[o];

  const float* xb = x + (size_t)b * CC * HH * WW;
  for (int c = 0; c < CC; ++c) {
    const float* xc = xb + (size_t)c * HH * WW;
#pragma unroll
    for (int ky = 0; ky < 3; ++ky) {
      int y = h + ky - 1;
      if (y < 0 || y >= HH) continue;
#pragma unroll
      for (int kx = 0; kx < 3; ++kx) {
        int xx = w + kx - 1;
        if (xx < 0 || xx >= WW) continue;
        float v = xc[y * WW + xx];
        int wi = c * K2 + ky * 3 + kx;
#pragma unroll
        for (int o = 0; o < OFFC; ++o) acc[o] += v * sw[o * (CC * K2) + wi];
      }
    }
  }
#pragma unroll
  for (int o = 0; o < OFFC; ++o)
    off[((size_t)b * OFFC + o) * HH * WW + pr] = acc[o];
}

// ---------------------------------------------------------------- kernel 4 --
// Deformable GEMM. Block = 32 pixels (one row segment) x 64 out channels.
//   phase 1: 288 bilinear coordinate sets -> LDS (indices + masked weights)
//   phase 2: cooperative gather of 32x576 sampled tile -> LDS bf16
//   phase 3: 8 waves x (16 outch x 16 pixel) WMMA tiles, K=576 in 18 steps
__global__ __launch_bounds__(256) void deform_gemm(
    const float*  __restrict__ x,     // fused x1*x2, fp32 [B][C][H][W]
    const float*  __restrict__ off,   // offsets fp32 [B][18][H][W]
    const __bf16* __restrict__ wbf,   // W_dc bf16 [64][576]
    const float*  __restrict__ x1,    // residual input
    const float*  __restrict__ bdc,   // bias [64]
    float*        __restrict__ out) {
  __shared__ float  s_w4[TP * K2][4];   // bilinear weights (0 if OOB)
  __shared__ int    s_i4[TP * K2][4];   // clamped gather indices (y*W+x)
  __shared__ __bf16 s_s[TP][SROW];      // sampled tile, bf16

  const int wtiles = WW / TP;           // 5
  int blk = blockIdx.x;
  int b   = blk / (HH * wtiles);
  int rem = blk % (HH * wtiles);
  int h   = rem / wtiles;
  int w0  = (rem % wtiles) * TP;
  int tid = threadIdx.x;

  const float* xb   = x   + (size_t)b * CC * HH * WW;
  const float* offb = off + (size_t)b * OFFC * HH * WW;

  // ---- phase 1: per-(pixel,kernel-tap) sampling coords -----------------
  for (int i = tid; i < TP * K2; i += 256) {
    int p = i / K2, kk = i % K2;
    int wpx = w0 + p;
    float dy = offb[(size_t)(kk * 2 + 0) * HH * WW + h * WW + wpx];
    float dx = offb[(size_t)(kk * 2 + 1) * HH * WW + h * WW + wpx];
    float py = (float)h   + (float)(kk / 3 - 1) + dy;
    float px = (float)wpx + (float)(kk % 3 - 1) + dx;
    float y0f = floorf(py), x0f = floorf(px);
    float wy1 = py - y0f, wx1 = px - x0f;
    float wy0 = 1.0f - wy1, wx0 = 1.0f - wx1;
    int y0 = (int)y0f, x0i = (int)x0f;
    float wgt[4] = {wy0 * wx0, wy0 * wx1, wy1 * wx0, wy1 * wx1};
#pragma unroll
    for (int j = 0; j < 4; ++j) {
      int yy = y0 + (j >> 1), xx = x0i + (j & 1);
      bool valid = (yy >= 0) && (yy < HH) && (xx >= 0) && (xx < WW);
      int yc = min(max(yy, 0), HH - 1);
      int xc = min(max(xx, 0), WW - 1);
      s_i4[i][j] = yc * WW + xc;
      s_w4[i][j] = valid ? wgt[j] : 0.0f;
    }
  }
  __syncthreads();

  // ---- phase 2: gather + bilinear blend -> bf16 LDS tile ---------------
  for (int i = tid; i < TP * KC; i += 256) {
    int p = i / KC, k = i % KC;
    int c = k / K2, kk = k % K2;
    int ci = p * K2 + kk;
    const float* xc = xb + (size_t)c * HH * WW;
    float v = s_w4[ci][0] * xc[s_i4[ci][0]] + s_w4[ci][1] * xc[s_i4[ci][1]] +
              s_w4[ci][2] * xc[s_i4[ci][2]] + s_w4[ci][3] * xc[s_i4[ci][3]];
    s_s[p][k] = (__bf16)v;
  }
  __syncthreads();

  // ---- phase 3: WMMA GEMM ---------------------------------------------
  int wid = tid >> 5, lane = tid & 31;
  int half = lane >> 4, ln = lane & 15;
  int mtile = wid & 3;    // out-channel tile (4 x 16 = 64)
  int pgrp  = wid >> 2;   // pixel group (2 x 16 = 32)

  // A lane layout (16-bit A 16x32): lane half h holds K = {h*8..h*8+7} and
  // {16+h*8..16+h*8+7} for row M = lane&15 -> two contiguous 16B loads.
  const __bf16* Arow = wbf + (size_t)(mtile * 16 + ln) * KC;
  // B lane layout (16-bit B 32x16): lane half h holds K = {h*8..h*8+7} and
  // {16+h*8..16+h*8+7} for column N = lane&15 -> two 16B LDS loads.
  const __bf16* Brow = &s_s[pgrp * 16 + ln][0];

  __builtin_prefetch(Arow, 0, 3);  // global_prefetch_b8 on L2-resident weights

  v8f acc = {};
#pragma unroll
  for (int ks = 0; ks < KC; ks += 32) {
    v8bf a0 = *(const v8bf*)(Arow + ks + half * 8);
    v8bf a1 = *(const v8bf*)(Arow + ks + 16 + half * 8);
    v8bf b0 = *(const v8bf*)(Brow + ks + half * 8);
    v8bf b1 = *(const v8bf*)(Brow + ks + 16 + half * 8);
    v16bf a, bm;
#pragma unroll
    for (int i = 0; i < 8; ++i) {
      a[i] = a0[i]; a[i + 8] = a1[i];
      bm[i] = b0[i]; bm[i + 8] = b1[i];
    }
    acc = __builtin_amdgcn_wmma_f32_16x16x32_bf16(
        /*neg_a=*/false, a, /*neg_b=*/false, bm,
        /*c_mod=*/(short)0, acc, /*reuse_a=*/false, /*reuse_b=*/false);
  }

  // C/D layout: lane half -> M offset 8, VGPR r -> row; N = lane&15 (pixel)
  int o_base = mtile * 16 + half * 8;
  int wpx = w0 + pgrp * 16 + ln;
#pragma unroll
  for (int r = 0; r < 8; ++r) {
    int o = o_base + r;
    size_t idx = (((size_t)b * CC + o) * HH + h) * WW + wpx;
    out[idx] = acc[r] + x1[idx] + bdc[o];
  }
}

// ---------------------------------------------------------------------------
extern "C" void kernel_launch(void* const* d_in, const int* in_sizes, int n_in,
                              void* d_out, int out_size, void* d_ws, size_t ws_size,
                              hipStream_t stream) {
  const float* x1   = (const float*)d_in[0];
  const float* x2   = (const float*)d_in[1];
  const float* woff = (const float*)d_in[2];
  const float* boff = (const float*)d_in[3];
  const float* wdc  = (const float*)d_in[4];
  const float* bdc  = (const float*)d_in[5];
  float* out = (float*)d_out;

  // workspace layout
  const size_t nX   = (size_t)BB * CC * HH * WW;         // 6,553,600 floats
  const size_t nOff = (size_t)BB * OFFC * HH * WW;       // 1,843,200 floats
  char* ws = (char*)d_ws;
  float*  x_fused = (float*)ws;                          // 26.2 MB
  float*  off     = (float*)(ws + nX * sizeof(float));   //  7.4 MB
  __bf16* wbf     = (__bf16*)(ws + (nX + nOff) * sizeof(float)); // 72 KB
  (void)in_sizes; (void)n_in; (void)out_size; (void)ws_size;

  // 1) x = x1 * x2
  int n4 = (int)(nX / 4);
  mul_kernel<<<(n4 + 255) / 256, 256, 0, stream>>>(x1, x2, x_fused, n4);

  // 2) W_dc -> bf16
  int nw = CC * KC;
  wconv_kernel<<<(nw + 255) / 256, 256, 0, stream>>>(wdc, wbf, nw);

  // 3) offset conv (18 channels)
  int npix = BB * HH * WW;
  offset_conv<<<(npix + 255) / 256, 256, 0, stream>>>(x_fused, woff, boff, off);

  // 4) deformable sampling + WMMA GEMM + residual
  int nblk = BB * HH * (WW / TP);                        // 3200 blocks
  deform_gemm<<<nblk, 256, 0, stream>>>(x_fused, off, wbf, x1, bdc, out);
}
